// Nonlocal_FC1_Class_13022340841601
// MI455X (gfx1250) — matrix operand order repaired and checked by reading, not compiled, not run on hardware
//
#include <hip/hip_runtime.h>
#include <hip/hip_bf16.h>

// ---------------- problem constants ----------------
#define BATCH 16
#define CIN   1024
#define DINT  512
#define NTOK  1568          // 8*14*14, = 49*32
#define NCLS  13
#define FS    1577          // f-strip LDS stride (odd -> conflict free columns)

typedef __attribute__((ext_vector_type(16))) __bf16        v16bf;
typedef __attribute__((ext_vector_type(8)))  float         v8f;
typedef __attribute__((ext_vector_type(4)))  unsigned int  v4u;
typedef __attribute__((ext_vector_type(8)))  unsigned int  v8u;

// round-to-nearest-even f32 -> bf16
__device__ inline unsigned short f2bf(float f) {
    unsigned u = __builtin_bit_cast(unsigned, f);
    u += 0x7FFFu + ((u >> 16) & 1u);
    return (unsigned short)(u >> 16);
}

// assemble a 16-element bf16 WMMA operand from two 16-byte LDS chunks
__device__ inline v16bf make_op(const unsigned short* p0, const unsigned short* p1) {
    v4u lo = *(const v4u*)p0;
    v4u hi = *(const v4u*)p1;
    v8u u;
    u[0] = lo[0]; u[1] = lo[1]; u[2] = lo[2]; u[3] = lo[3];
    u[4] = hi[0]; u[5] = hi[1]; u[6] = hi[2]; u[7] = hi[3];
    return __builtin_bit_cast(v16bf, u);
}

// generic __shared__ pointer -> 32-bit LDS byte offset (flat aperture: low 32 bits)
__device__ inline unsigned lds_addr(const void* p) {
    return (unsigned)(unsigned long long)p;
}

// CDNA5 async global->LDS 16-byte copy (ASYNCcnt-tracked, no VGPR data path)
__device__ inline void async_b128(unsigned lds_byte, const void* gptr) {
    asm volatile("global_load_async_to_lds_b128 %0, %1, off"
                 :: "v"(lds_byte), "v"(gptr) : "memory");
}

// ---------------- kernel 0: zero cbar accumulator ----------------
__global__ void zero_kernel(float* p, int n) {
    int i = blockIdx.x * blockDim.x + threadIdx.x;
    if (i < n) p[i] = 0.0f;
}

// ---------------- kernel 1: theta/phi projections -> bf16 [B,N,512] ----------------
// grid = (49, 16, 16): z>>3 selects theta/phi, (z&7)*64 = output-channel base
__launch_bounds__(256)
__global__ void proj_kernel(const float* __restrict__ x,
                            const float* __restrict__ thw, const float* __restrict__ thb,
                            const float* __restrict__ phw, const float* __restrict__ phb,
                            unsigned short* __restrict__ TH, unsigned short* __restrict__ PH)
{
    const int nBase = blockIdx.x * 32;
    const int b     = blockIdx.y;
    const int z     = blockIdx.z;
    const int which = z >> 3;
    const int oBase = (z & 7) * 64;
    const float* w    = which ? phw : thw;
    const float* bias = which ? phb : thb;
    unsigned short* out = which ? PH : TH;

    __shared__ unsigned short sX[32 * 72];   // x^T tile: [n][c], kTile=64, padded stride
    __shared__ unsigned short sW[64 * 72];   // weight tile: [o][c]

    const int t    = threadIdx.x;
    const int lane = t & 31;
    const int wid  = t >> 5;
    const int wn   = wid & 1;       // n 16-tile (2)
    const int wo   = wid >> 1;      // o 16-tile (4)
    const int l15  = lane & 15;
    const int hi16 = lane >> 4;

    v8f acc = {0.f,0.f,0.f,0.f,0.f,0.f,0.f,0.f};

    for (int k0 = 0; k0 < CIN; k0 += 64) {
        __syncthreads();
        {   // x tile: c in [k0,k0+64), n in [nBase,nBase+32); store transposed as bf16
            int c  = t >> 2;
            int n0 = (t & 3) * 8;
            const float* src = x + (((size_t)b * CIN + k0 + c) * NTOK + nBase + n0);
            if (k0 + 64 < CIN) __builtin_prefetch(src + (size_t)64 * NTOK, 0, 2);
            float4 v0 = *(const float4*)(src);
            float4 v1 = *(const float4*)(src + 4);
            unsigned short h[8] = { f2bf(v0.x), f2bf(v0.y), f2bf(v0.z), f2bf(v0.w),
                                    f2bf(v1.x), f2bf(v1.y), f2bf(v1.z), f2bf(v1.w) };
            #pragma unroll
            for (int i = 0; i < 8; ++i) sX[(n0 + i) * 72 + c] = h[i];
        }
        {   // weight tile: o in [oBase,+64), c in [k0,+64)
            int o  = t >> 2;
            int c0 = (t & 3) * 16;
            const float* src = w + ((size_t)(oBase + o) * CIN + k0 + c0);
            if (k0 + 64 < CIN) __builtin_prefetch(src + 64, 0, 2);
            float4 v0 = *(const float4*)(src + 0);
            float4 v1 = *(const float4*)(src + 4);
            float4 v2 = *(const float4*)(src + 8);
            float4 v3 = *(const float4*)(src + 12);
            uint4 s0, s1;
            s0.x = f2bf(v0.x) | ((unsigned)f2bf(v0.y) << 16);
            s0.y = f2bf(v0.z) | ((unsigned)f2bf(v0.w) << 16);
            s0.z = f2bf(v1.x) | ((unsigned)f2bf(v1.y) << 16);
            s0.w = f2bf(v1.z) | ((unsigned)f2bf(v1.w) << 16);
            s1.x = f2bf(v2.x) | ((unsigned)f2bf(v2.y) << 16);
            s1.y = f2bf(v2.z) | ((unsigned)f2bf(v2.w) << 16);
            s1.z = f2bf(v3.x) | ((unsigned)f2bf(v3.y) << 16);
            s1.w = f2bf(v3.z) | ((unsigned)f2bf(v3.w) << 16);
            unsigned short* dst = &sW[o * 72 + c0];
            *(uint4*)(dst)     = s0;
            *(uint4*)(dst + 8) = s1;
        }
        __syncthreads();
        #pragma unroll
        for (int kk = 0; kk < 64; kk += 32) {
            // A (weights, M=o): lanes<16 K{kk..kk+7, kk+16..23}; lanes>=16 +8
            const unsigned short* ap = &sW[(wo * 16 + l15) * 72 + kk + hi16 * 8];
            v16bf a = make_op(ap, ap + 16);
            // B (x^T, N=n): lanes<16 K{kk..kk+7, kk+16..23}; lanes>=16 +8
            const unsigned short* bp = &sX[(wn * 16 + l15) * 72 + kk + hi16 * 8];
            v16bf bb = make_op(bp, bp + 16);
            acc = __builtin_amdgcn_wmma_f32_16x16x32_bf16(false, a, false, bb,
                                                          (short)0, acc, false, false);
        }
    }

    // epilogue: +bias, pack 8 bf16, one 16B store per lane
    const int oo = oBase + wo * 16 + hi16 * 8;
    const int n  = nBase + wn * 16 + l15;
    unsigned short h[8];
    #pragma unroll
    for (int i = 0; i < 8; ++i) h[i] = f2bf(acc[i] + bias[oo + i]);
    uint4 st;
    st.x = h[0] | ((unsigned)h[1] << 16);
    st.y = h[2] | ((unsigned)h[3] << 16);
    st.z = h[4] | ((unsigned)h[5] << 16);
    st.w = h[6] | ((unsigned)h[7] << 16);
    *(uint4*)(out + (((size_t)b * NTOK + n) * DINT + oo)) = st;
}

// ---------------- kernel 2: f = TH*PH^T strip, softmax, column means -> cbar ----------------
// grid = (49, 16); dynamic LDS holds the whole 32 x 1568 f32 strip.
// PH tiles are double-buffered and moved with global_load_async_to_lds_b128.
#define NTILE 200   // 25 mBase steps * 8 k0 steps

__device__ __forceinline__ void issue_ph(const unsigned short* __restrict__ PH,
                                         int b, int t, int mBase, int k0,
                                         unsigned short* buf)
{
    #pragma unroll
    for (int j = 0; j < 2; ++j) {           // 512 x 16B chunks over 256 threads
        int u   = t + j * 256;
        int row = u >> 3, col = u & 7;
        int m   = mBase + row;
        if (m > NTOK - 1) m = NTOK - 1;     // clamp: keeps exactly 2 issues/thread/tile
        async_b128(lds_addr(buf + row * 136 + col * 8),
                   PH + (((size_t)b * NTOK + m) * DINT) + k0 + col * 8);
    }
}

__launch_bounds__(256)
__global__ void attn_kernel(const unsigned short* __restrict__ TH,
                            const unsigned short* __restrict__ PH,
                            float* __restrict__ cbar)
{
    extern __shared__ char smem[];
    float*          fS  = (float*)smem;                              // 32*FS f32
    unsigned short* thS = (unsigned short*)(smem + 32 * FS * 4);     // 32 rows * 520
    unsigned short* ph0 = thS + 32 * 520;                            // 64 rows * 136
    unsigned short* ph1 = ph0 + 64 * 136;                            // double buffer
    float*          rst = (float*)(ph1 + 64 * 136);                  // 64 f32 stats

    const int b     = blockIdx.y;
    const int nBase = blockIdx.x * 32;
    const int t     = threadIdx.x;
    const int lane  = t & 31;
    const int wid   = t >> 5;
    const int wn    = wid & 1;     // n 16-tile (2)
    const int wm    = wid >> 1;    // m 16-tile (4)
    const int l15   = lane & 15;
    const int hi16  = lane >> 4;

    // stage TH strip (all K=512) asynchronously, 8 chunks per thread
    for (int u = t; u < 2048; u += 256) {
        int row = u >> 6, col = u & 63;
        async_b128(lds_addr(thS + row * 520 + col * 8),
                   TH + (((size_t)b * NTOK + nBase + row) * DINT) + col * 8);
    }
    // prologue: PH tile 0 into buffer 0
    issue_ph(PH, b, t, 0, 0, ph0);

    const v8f vzero = {0.f,0.f,0.f,0.f,0.f,0.f,0.f,0.f};
    v8f acc = vzero;

    for (int tt = 0; tt < NTILE; ++tt) {
        unsigned short* cur = (tt & 1) ? ph1 : ph0;
        unsigned short* nxt = (tt & 1) ? ph0 : ph1;
        if (tt + 1 < NTILE) {
            const int t2 = tt + 1;
            issue_ph(PH, b, t, (t2 >> 3) * 64, (t2 & 7) * 64, nxt);
            // own wave: 2 outstanding left == tile tt complete (async loads retire in order)
            asm volatile("s_wait_asynccnt 0x2" ::: "memory");
        } else {
            asm volatile("s_wait_asynccnt 0x0" ::: "memory");
        }
        __syncthreads();   // all waves' async data for tile tt now visible

        const int mBase = (tt >> 3) * 64;
        const int k0    = (tt & 7) * 64;
        if ((tt & 7) == 0) acc = vzero;

        #pragma unroll
        for (int kk = 0; kk < 64; kk += 32) {
            const unsigned short* ap = thS + (wn * 16 + l15) * 520 + k0 + kk + hi16 * 8;
            v16bf a = make_op(ap, ap + 16);
            const unsigned short* bp = cur + (wm * 16 + l15) * 136 + kk + hi16 * 8;
            v16bf bb = make_op(bp, bp + 16);
            acc = __builtin_amdgcn_wmma_f32_16x16x32_bf16(false, a, false, bb,
                                                          (short)0, acc, false, false);
        }

        if ((tt & 7) == 7) {
            int mcol = mBase + wm * 16 + l15;
            if (mcol < NTOK) {
                #pragma unroll
                for (int i = 0; i < 8; ++i)
                    fS[(wn * 16 + i + hi16 * 8) * FS + mcol] = acc[i];
            }
        }
        __syncthreads();   // protect 'cur' before it is refilled next iteration
    }

    // row-wise softmax stats over the strip (8 threads per row)
    float* rmax = rst;
    float* rinv = rst + 32;
    const int r = t >> 3, s = t & 7;
    float mx = -1e30f;
    for (int m = s; m < NTOK; m += 8) mx = fmaxf(mx, fS[r * FS + m]);
    for (int o = 4; o >= 1; o >>= 1) mx = fmaxf(mx, __shfl_xor(mx, o, 8));
    if (s == 0) rmax[r] = mx;
    __syncthreads();
    const float mxr = rmax[r];
    float sum = 0.f;
    for (int m = s; m < NTOK; m += 8) {
        float e = __expf(fS[r * FS + m] - mxr);
        fS[r * FS + m] = e;
        sum += e;
    }
    for (int o = 4; o >= 1; o >>= 1) sum += __shfl_xor(sum, o, 8);
    if (s == 0) rinv[r] = 1.0f / sum;
    __syncthreads();

    // column partial sums of attn over this strip -> global accumulate
    for (int m = t; m < NTOK; m += 256) {
        float a = 0.f;
        #pragma unroll 4
        for (int rr = 0; rr < 32; ++rr) a += fS[rr * FS + m] * rinv[rr];
        atomicAdd(&cbar[(size_t)b * NTOK + m], a);
    }
}

// ---------------- kernel 3: xc = x.cbar , xbar = mean_n x ----------------
__launch_bounds__(256)
__global__ void reduce_kernel(const float* __restrict__ x, const float* __restrict__ cbar,
                              float* __restrict__ xc, float* __restrict__ xbar)
{
    __shared__ float scb[NTOK];
    const int b = blockIdx.y;
    const int cBase = blockIdx.x * 64;
    const int t = threadIdx.x;
    for (int m = t; m < NTOK; m += 256) scb[m] = cbar[(size_t)b * NTOK + m];
    __syncthreads();
    const int c = cBase + (t >> 2), q = t & 3;
    const float* xr = x + ((size_t)b * CIN + c) * NTOK;
    float dot = 0.f, sm = 0.f;
    for (int m = q * 4; m < NTOK; m += 16) {
        float4 v = *(const float4*)(xr + m);
        dot += v.x * scb[m] + v.y * scb[m + 1] + v.z * scb[m + 2] + v.w * scb[m + 3];
        sm  += v.x + v.y + v.z + v.w;
    }
    for (int o = 2; o >= 1; o >>= 1) { dot += __shfl_xor(dot, o, 4); sm += __shfl_xor(sm, o, 4); }
    if (q == 0) {
        xc[(size_t)b * CIN + c]   = dot;
        xbar[(size_t)b * CIN + c] = sm * (1.0f / NTOK);
    }
}

// ---------------- kernel 4: head: ybar -> pooled -> FC ----------------
__launch_bounds__(256)
__global__ void head_kernel(const float* __restrict__ gw, const float* __restrict__ gb,
                            const float* __restrict__ ww, const float* __restrict__ wb,
                            const float* __restrict__ fcw, const float* __restrict__ fcb,
                            const float* __restrict__ xc, const float* __restrict__ xbar,
                            float* __restrict__ out)
{
    __shared__ float sxc[CIN];
    __shared__ float syb[DINT];
    __shared__ float spl[CIN];
    const int b = blockIdx.x, t = threadIdx.x;
    for (int c = t; c < CIN; c += 256) sxc[c] = xc[(size_t)b * CIN + c];
    __syncthreads();
    for (int d = t; d < DINT; d += 256) {
        const float* wr = gw + (size_t)d * CIN;
        float acc = 0.f;
        for (int c = 0; c < CIN; c += 4) {
            float4 v = *(const float4*)(wr + c);
            acc += v.x * sxc[c] + v.y * sxc[c + 1] + v.z * sxc[c + 2] + v.w * sxc[c + 3];
        }
        syb[d] = acc * (1.0f / NTOK) + gb[d];
    }
    __syncthreads();
    for (int c = t; c < CIN; c += 256) {
        const float* wr = ww + (size_t)c * DINT;
        float acc = 0.f;
        for (int d = 0; d < DINT; d += 4) {
            float4 v = *(const float4*)(wr + d);
            acc += v.x * syb[d] + v.y * syb[d + 1] + v.z * syb[d + 2] + v.w * syb[d + 3];
        }
        spl[c] = acc + wb[c] + xbar[(size_t)b * CIN + c];
    }
    __syncthreads();
    if (t < NCLS) {
        const float* wr = fcw + (size_t)t * CIN;
        float acc = 0.f;
        for (int c = 0; c < CIN; ++c) acc += wr[c] * spl[c];
        out[b * NCLS + t] = acc + fcb[t];
    }
}

// ---------------- launch ----------------
extern "C" void kernel_launch(void* const* d_in, const int* in_sizes, int n_in,
                              void* d_out, int out_size, void* d_ws, size_t ws_size,
                              hipStream_t stream)
{
    (void)in_sizes; (void)n_in; (void)out_size; (void)ws_size;
    const float* x    = (const float*)d_in[0];
    const float* g_w  = (const float*)d_in[1];
    const float* g_b  = (const float*)d_in[2];
    const float* th_w = (const float*)d_in[3];
    const float* th_b = (const float*)d_in[4];
    const float* ph_w = (const float*)d_in[5];
    const float* ph_b = (const float*)d_in[6];
    const float* w_w  = (const float*)d_in[7];
    const float* w_b  = (const float*)d_in[8];
    const float* fc_w = (const float*)d_in[9];
    const float* fc_b = (const float*)d_in[10];
    float* out = (float*)d_out;

    // workspace layout (bf16 TH/PH, f32 accumulators); total ~49.3 MiB
    unsigned short* TH = (unsigned short*)d_ws;
    unsigned short* PH = TH + (size_t)BATCH * NTOK * DINT;
    float* cbar = (float*)(PH + (size_t)BATCH * NTOK * DINT);
    float* xc   = cbar + (size_t)BATCH * NTOK;
    float* xbar = xc   + (size_t)BATCH * CIN;

    zero_kernel<<<dim3((BATCH * NTOK + 255) / 256), 256, 0, stream>>>(cbar, BATCH * NTOK);

    proj_kernel<<<dim3(NTOK / 32, BATCH, 16), 256, 0, stream>>>(
        x, th_w, th_b, ph_w, ph_b, TH, PH);

    // fS + thS + 2*phS + stats = 201856 + 33280 + 34816 + 256 bytes (< 320KB/WGP)
    size_t smem = (size_t)32 * FS * 4 + (size_t)32 * 520 * 2 + 2 * (size_t)64 * 136 * 2 + 256;
    attn_kernel<<<dim3(NTOK / 32, BATCH), 256, smem, stream>>>(TH, PH, cbar);

    reduce_kernel<<<dim3(CIN / 64, BATCH), 256, 0, stream>>>(x, cbar, xc, xbar);

    head_kernel<<<BATCH, 256, 0, stream>>>(g_w, g_b, w_w, w_b, fc_w, fc_b, xc, xbar, out);
}
